// FMNLayer_41901700940050
// MI455X (gfx1250) — matrix operand order repaired and checked by reading, not compiled, not run on hardware
//
#include <hip/hip_runtime.h>
#include <hip/hip_bf16.h>

#define BATCH 16
#define SEQ   2048
#define DIM   512
#define DBD   128
#define KSLOT 7
#define NTOP  16
#define WIN   64
#define KC    16            // DIM / 32 k-steps
#define NEGF  (-1e30f)

static __device__ __forceinline__ float PHI_F() { return 1.618033988749895f; }
static __device__ __forceinline__ float PSI_F() { return -0.6180339887498949f; }

typedef __attribute__((ext_vector_type(16))) __bf16 v16bf;
typedef __attribute__((ext_vector_type(8)))  float  v8f;
typedef __attribute__((ext_vector_type(4)))  unsigned int u32x4_t;
typedef __attribute__((ext_vector_type(8)))  int i32x8_t;
typedef __attribute__((ext_vector_type(4)))  int i32x4_t;

#if defined(__has_builtin)
#if __has_builtin(__builtin_amdgcn_tensor_load_to_lds) && __has_builtin(__builtin_amdgcn_s_wait_tensorcnt)
#define HAVE_TDM 1
#endif
#endif
#ifndef HAVE_TDM
#define HAVE_TDM 0
#endif

// ---- compile-time probes (visible in stderr; compile-only feedback loop) ----
#define PROBE_STR2(x) #x
#define PROBE_STR(x) PROBE_STR2(x)
#pragma message("CDNA5 probe: __clang_major__ = " PROBE_STR(__clang_major__))
#if HAVE_TDM
#warning "CDNA5 probe: tensor_load_to_lds builtin path ENABLED (TDM staging active)"
#else
#warning "CDNA5 probe: tensor_load_to_lds builtin NOT available -> vector-copy fallback"
#endif

__device__ __forceinline__ unsigned short bfbits(float f) {
  union { float f; unsigned u; } a; a.f = f;
  unsigned r = a.u + 0x7FFFu + ((a.u >> 16) & 1u);   // round-to-nearest-even
  return (unsigned short)(r >> 16);
}
__device__ __forceinline__ __bf16 f2bf(float f) {
  union { unsigned short s; __bf16 b; } o; o.s = bfbits(f);
  return o.b;
}

// A-fragment tiled offset for element (row, k) of a [rows x DIM] matrix.
// Per 16x32 subtile: lane = (row&15) + 16*bit4(k), pos = k&15; 16 contiguous per lane.
__device__ __forceinline__ size_t a_tiled_off(size_t row, int k) {
  size_t rt = row >> 4;
  int kc = k >> 5;
  int lane = (int)(row & 15) + 16 * ((k >> 4) & 1);
  int pos = k & 15;
  return ((rt * KC + kc) * 32 + lane) * 16 + pos;
}

#if HAVE_TDM
// 2D tensor-DMA: nrows rows of row_elems bf16 (src stride = stride_elems), packed
// contiguously into LDS. Issue from ONE wave; TENSORcnt waited by the issuing wave.
__device__ __forceinline__ void tdm_load_2d(const void* gsrc, void* ldst,
                                            unsigned row_elems, unsigned nrows,
                                            unsigned stride_elems) {
  unsigned long long ga = (unsigned long long)(uintptr_t)gsrc;
  unsigned lds_off = (unsigned)(uintptr_t)ldst;
  u32x4_t g0;
  g0[0] = 1u;                                            // count=1 valid descriptor
  g0[1] = lds_off;                                       // lds_addr (bytes)
  g0[2] = (unsigned)ga;                                  // global_addr[31:0]
  g0[3] = ((unsigned)(ga >> 32) & 0x01FFFFFFu) | 0x80000000u;  // addr[56:32] | type=2
  i32x8_t g1;
  g1[0] = 1 << 16;                                       // data_size = 1 (2 bytes)
  g1[1] = (int)((stride_elems & 0xFFFFu) << 16);         // tensor_dim0 low16
  g1[2] = (int)((stride_elems >> 16) | ((nrows & 0xFFFFu) << 16)); // td0 hi | td1 lo
  g1[3] = (int)(((row_elems & 0xFFFFu) << 16) | (nrows >> 16));    // tile_dim0 | td1 hi
  g1[4] = (int)(nrows & 0xFFFFu);                        // tile_dim1
  g1[5] = (int)stride_elems;                             // tensor_dim0_stride low32
  g1[6] = 0;
  g1[7] = 0;
  i32x4_t gz = {0, 0, 0, 0};
#if __clang_major__ >= 23
  i32x8_t gz8 = {0, 0, 0, 0, 0, 0, 0, 0};
  __builtin_amdgcn_tensor_load_to_lds(g0, g1, gz, gz, gz8, 0);
#else
  __builtin_amdgcn_tensor_load_to_lds(g0, g1, gz, gz, 0);
#endif
}
#endif

// ---------------- LayerNorm of X_write -> xn (row-major f32) + xnt (tiled bf16) --------
__global__ void k_ln(const float* __restrict__ X, const float* __restrict__ w,
                     const float* __restrict__ bia, float* __restrict__ xn,
                     __bf16* __restrict__ xnt) {
  size_t row = blockIdx.x;
  const float* x = X + row * DIM;
  int tid = threadIdx.x;
  float v0 = x[tid], v1 = x[tid + 256];
  __shared__ float red[256];
  red[tid] = v0 + v1; __syncthreads();
  for (int st = 128; st > 0; st >>= 1) { if (tid < st) red[tid] += red[tid + st]; __syncthreads(); }
  float mu = red[0] / (float)DIM; __syncthreads();
  float d0 = v0 - mu, d1 = v1 - mu;
  red[tid] = d0 * d0 + d1 * d1; __syncthreads();
  for (int st = 128; st > 0; st >>= 1) { if (tid < st) red[tid] += red[tid + st]; __syncthreads(); }
  float rstd = rsqrtf(red[0] / (float)DIM + 1e-5f);
  float y0 = d0 * rstd * w[tid] + bia[tid];
  float y1 = d1 * rstd * w[tid + 256] + bia[tid + 256];
  xn[row * DIM + tid] = y0;       xn[row * DIM + tid + 256] = y1;
  xnt[a_tiled_off(row, tid)] = f2bf(y0);
  xnt[a_tiled_off(row, tid + 256)] = f2bf(y1);
}

// ---------------- f32 -> A-fragment-tiled bf16 ----------------
__global__ void k_cvt_tiled(const float* __restrict__ x, __bf16* __restrict__ yt) {
  size_t n = (size_t)BATCH * SEQ * DIM;
  size_t i = (size_t)blockIdx.x * blockDim.x + threadIdx.x;
  size_t stride = (size_t)gridDim.x * blockDim.x;
  for (; i < n; i += stride) {
    size_t row = i >> 9;
    int k = (int)(i & 511);
    yt[a_tiled_off(row, k)] = f2bf(x[i]);
  }
}

// ---------------- weight pre-tiling into B-fragment layout ----------------
// B[k][n] fragment: lane = (n&15)+16*bit4(k), pos = k&15 -> 16 contiguous per lane.
// TRANS=1: B[k][n] = Wsrc[n*DIM + k]; TRANS=0: B[k][n] = Wsrc[k*N + n].
template<int TRANS>
__global__ void k_tileB(const float* __restrict__ Wsrc, __bf16* __restrict__ Bt, int Ntiles) {
  size_t total = (size_t)Ntiles * KC * 512;
  size_t gid = (size_t)blockIdx.x * blockDim.x + threadIdx.x;
  if (gid >= total) return;
  int pos = (int)(gid & 15);
  int lane = (int)((gid >> 4) & 31);
  int kc = (int)((gid >> 9) & 15);
  int nt = (int)(gid >> 13);
  int n = nt * 16 + (lane & 15);
  int k = kc * 32 + ((lane >> 4) & 1) * 16 + pos;
  float v = TRANS ? Wsrc[(size_t)n * DIM + k] : Wsrc[(size_t)k * (Ntiles * 16) + n];
  Bt[gid] = f2bf(v);
}

// ---------------- bf16 WMMA GEMM with TDM-staged A panel -------------------------------
// C[M,64-per-blockIdx.y] ; A tiled (At), B tiled (Bt), both fragment layout.
__global__ void k_gemm(const __bf16* __restrict__ At, const __bf16* __restrict__ Bt,
                       float* __restrict__ C, int Ntot) {
  __shared__ __attribute__((aligned(64))) __bf16 Apanel[4 * 8 * 32 * 16]; // 32 KB
  const int tid = threadIdx.x;
  const int m0 = blockIdx.x * 64;
  const int n0 = blockIdx.y * 64;
  const int wave = tid >> 5, lane = tid & 31;
  const int half = lane >> 4, lmod = lane & 15;
  const int tr = (wave * 2) >> 2;
  const int tc0 = (wave * 2) & 3, tc1 = (wave * 2 + 1) & 3;
  const int ntg0 = (n0 >> 4) + tc0, ntg1 = (n0 >> 4) + tc1;
  v8f acc0 = {0.f,0.f,0.f,0.f,0.f,0.f,0.f,0.f};
  v8f acc1 = {0.f,0.f,0.f,0.f,0.f,0.f,0.f,0.f};
  const size_t abase = (size_t)(m0 >> 4) * KC * 512;
  for (int kp = 0; kp < KC; kp += 8) {
    __syncthreads();
#if HAVE_TDM
    if (tid < 32) {
      tdm_load_2d(At + abase + (size_t)kp * 512, Apanel,
                  /*row_elems=*/8 * 512, /*nrows=*/4, /*stride=*/KC * 512);
      __builtin_amdgcn_s_wait_tensorcnt(0);
    }
#else
    for (int idx = tid; idx < 2048; idx += 256) {
      int r = idx >> 9, cchunk = idx & 511;
      const __bf16* src = At + abase + (size_t)r * KC * 512 + (size_t)kp * 512 + cchunk * 8;
      *(uint4*)&Apanel[r * 4096 + cchunk * 8] = *(const uint4*)src;
    }
#endif
    __syncthreads();
    for (int kc2 = 0; kc2 < 8; ++kc2) {
      int kc = kp + kc2;
      v16bf af  = *(const v16bf*)&Apanel[((tr * 8 + kc2) * 32 + lane) * 16];
      v16bf bf0 = *(const v16bf*)(Bt + ((size_t)(ntg0 * KC + kc) * 32 + lane) * 16);
      v16bf bf1 = *(const v16bf*)(Bt + ((size_t)(ntg1 * KC + kc) * 32 + lane) * 16);
      if (kc + 1 < KC)
        __builtin_prefetch(Bt + ((size_t)(ntg0 * KC + kc + 1) * 32) * 16, 0, 1);
      acc0 = __builtin_amdgcn_wmma_f32_16x16x32_bf16(false, af, false, bf0, (short)0, acc0, false, false);
      acc1 = __builtin_amdgcn_wmma_f32_16x16x32_bf16(false, af, false, bf1, (short)0, acc1, false, false);
    }
  }
  #pragma unroll
  for (int v = 0; v < 8; ++v) {
    int row = v + 8 * half;
    int gr = m0 + tr * 16 + row;
    C[(size_t)gr * Ntot + n0 + tc0 * 16 + lmod] = acc0[v];
    C[(size_t)gr * Ntot + n0 + tc1 * 16 + lmod] = acc1[v];
  }
}

// ---------------- banded salience via WMMA: sal[b, i*WIN + (o-1)] ----------------
__global__ void k_sal(const __bf16* __restrict__ xnt, const float* __restrict__ xw0,
                      float* __restrict__ sal) {
  __shared__ __attribute__((aligned(32))) __bf16 BsF[5 * 32 * 16];
  const int b = blockIdx.y;
  const int i0 = blockIdx.x * 16;
  const int j0 = i0 - WIN;
  const int tid = threadIdx.x;
  const int wave = tid >> 5, lane = tid & 31;
  const int half = lane >> 4, lmod = lane & 15;
  const size_t rtg = ((size_t)b * SEQ + i0) >> 4;
  v8f acc = {0.f,0.f,0.f,0.f,0.f,0.f,0.f,0.f};
  for (int kc = 0; kc < KC; ++kc) {
    __syncthreads();
    for (int idx = tid; idx < 320; idx += 256) {
      int n = idx >> 2, c = idx & 3;
      int j = j0 + n;
      float4 f0 = make_float4(0.f, 0.f, 0.f, 0.f), f1 = f0;
      if (j >= 0) {
        const float* src = xw0 + ((size_t)b * SEQ + j) * DIM + kc * 32 + c * 8;
        f0 = *(const float4*)src;
        f1 = *(const float4*)(src + 4);
      }
      uint4 pk;
      pk.x = (unsigned)bfbits(f0.x) | ((unsigned)bfbits(f0.y) << 16);
      pk.y = (unsigned)bfbits(f0.z) | ((unsigned)bfbits(f0.w) << 16);
      pk.z = (unsigned)bfbits(f1.x) | ((unsigned)bfbits(f1.y) << 16);
      pk.w = (unsigned)bfbits(f1.z) | ((unsigned)bfbits(f1.w) << 16);
      int tc = n >> 4;
      int lane2 = (n & 15) + 16 * (c >> 1);
      int posb = (c & 1) * 8;
      *(uint4*)&BsF[(tc * 32 + lane2) * 16 + posb] = pk;
    }
    __syncthreads();
    if (wave < 5) {
      v16bf af  = *(const v16bf*)(xnt + ((rtg * KC + kc) * 32 + lane) * 16);
      v16bf bfr = *(const v16bf*)&BsF[(wave * 32 + lane) * 16];
      acc = __builtin_amdgcn_wmma_f32_16x16x32_bf16(false, af, false, bfr, (short)0, acc, false, false);
    }
  }
  if (wave < 5) {
    #pragma unroll
    for (int v = 0; v < 8; ++v) {
      int i = i0 + v + 8 * half;
      int j = j0 + wave * 16 + lmod;
      int o = i - j;
      if (o >= 1 && o <= WIN) {
        float val = (j >= 0) ? acc[v] : NEGF;
        sal[((size_t)b * SEQ + i) * WIN + (o - 1)] = val;
      }
    }
  }
}

// ---------------- per-batch top-16 (deterministic, lower-index tie-break) ----------------
__global__ void k_topk(const float* __restrict__ sal, float* __restrict__ tkv,
                       int* __restrict__ tki) {
  const int b = blockIdx.x;
  const float* s = sal + (size_t)b * SEQ * WIN;
  const int N = SEQ * WIN;
  __shared__ float sv[256];
  __shared__ int si[256];
  __shared__ int chosen[NTOP];
  for (int it = 0; it < NTOP; ++it) {
    float best = -3e38f; int bi = 0x7fffffff;
    for (int n = threadIdx.x; n < N; n += 256) {
      bool skip = false;
      for (int c = 0; c < it; ++c) if (chosen[c] == n) skip = true;
      float v = s[n];
      if (!skip && (v > best || (v == best && n < bi))) { best = v; bi = n; }
    }
    sv[threadIdx.x] = best; si[threadIdx.x] = bi; __syncthreads();
    for (int st = 128; st > 0; st >>= 1) {
      if (threadIdx.x < st) {
        float ov = sv[threadIdx.x + st]; int oi = si[threadIdx.x + st];
        if (ov > sv[threadIdx.x] || (ov == sv[threadIdx.x] && oi < si[threadIdx.x])) {
          sv[threadIdx.x] = ov; si[threadIdx.x] = oi;
        }
      }
      __syncthreads();
    }
    if (threadIdx.x == 0) {
      chosen[it] = si[0];
      tkv[b * NTOP + it] = sv[0];
      tki[b * NTOP + it] = si[0];
    }
    __syncthreads();
  }
}

// ---------------- slot write + strand mixer + losses (one block per batch) ----------------
__global__ void k_slots(const float* __restrict__ xn, const float* __restrict__ tkv,
                        const int* __restrict__ tki, const float* __restrict__ slots,
                        const float* __restrict__ W_V, const float* __restrict__ W_K_slot,
                        const float* __restrict__ skb, const float* __restrict__ Wg,
                        const float* __restrict__ Wgb, const float* __restrict__ lnw,
                        const float* __restrict__ lnb, const float* __restrict__ WsQ,
                        const float* __restrict__ WsK, const float* __restrict__ WsV,
                        const float* __restrict__ lam, const float* __restrict__ temps,
                        float* __restrict__ sn_ws, float* __restrict__ out_slots,
                        float* __restrict__ lossP, float* __restrict__ lossD) {
  const int b = blockIdx.x;
  const int d = threadIdx.x; // 0..127
  __shared__ int ii[NTOP], jj[NTOP];
  __shared__ float pv[NTOP], alpha[NTOP];
  __shared__ float Rt[DBD], red[DBD];
  __shared__ float compat[KSLOT], sw[KSLOT];
  __shared__ float supd[KSLOT][DBD], Bn[KSLOT][DBD];
  __shared__ float Qm[KSLOT][DBD], Km[KSLOT][DBD], Vm[KSLOT][DBD];
  __shared__ float sa[KSLOT][KSLOT];
  __shared__ float snew[KSLOT][DBD];
  __shared__ float norms[KSLOT], cosv[KSLOT * KSLOT];

  if (d < NTOP) {
    int f = tki[b * NTOP + d];
    int i = f / WIN;
    int o = (f % WIN) + 1;
    int j = i - o;
    ii[d] = i; jj[d] = (j < 0) ? 0 : j;
    pv[d] = tkv[b * NTOP + d];
  }
  __syncthreads();
  if (d == 0) {
    float mx = pv[0];
    for (int k = 1; k < NTOP; ++k) mx = fmaxf(mx, pv[k]);
    float s = 0.f;
    for (int k = 0; k < NTOP; ++k) { alpha[k] = expf(pv[k] - mx); s += alpha[k]; }
    for (int k = 0; k < NTOP; ++k) alpha[k] /= s;
  }
  __syncthreads();
  {
    float acc = 0.f;
    for (int k = 0; k < NTOP; ++k) {
      const float* xi = xn + ((size_t)b * SEQ + ii[k]) * DIM;
      const float* xj = xn + ((size_t)b * SEQ + jj[k]) * DIM;
      float r = 0.f;
      for (int c = 0; c < DIM; ++c) r += (xi[c] - xj[c]) * W_V[(size_t)c * DBD + d];
      acc += alpha[k] * r;
    }
    Rt[d] = acc;
  }
  __syncthreads();
  const float scale = rsqrtf((float)DBD);
  for (int k = 0; k < KSLOT; ++k) {
    const float* sl = slots + ((size_t)b * KSLOT + k) * DBD;
    float sk = skb[k * DBD + d];
    for (int c = 0; c < DBD; ++c) sk += sl[c] * W_K_slot[c * DBD + d];
    red[d] = sk * Rt[d]; __syncthreads();
    for (int st = 64; st > 0; st >>= 1) { if (d < st) red[d] += red[d + st]; __syncthreads(); }
    if (d == 0) {
      float sp = logf(1.f + expf(temps[k]));
      compat[k] = red[0] * scale * sp;
    }
    __syncthreads();
  }
  if (d == 0) {
    float mx = compat[0];
    for (int k = 1; k < KSLOT; ++k) mx = fmaxf(mx, compat[k]);
    float s = 0.f;
    for (int k = 0; k < KSLOT; ++k) { sw[k] = expf(compat[k] - mx); s += sw[k]; }
    for (int k = 0; k < KSLOT; ++k) sw[k] /= s;
  }
  __syncthreads();
  for (int k = 0; k < KSLOT; ++k) {
    const float* sl = slots + ((size_t)b * KSLOT + k) * DBD;
    float gi = Wgb[d];
    for (int c = 0; c < DBD; ++c) gi += sl[c] * Wg[c * DBD + d];
    float swk = sw[k];
    for (int c = 0; c < DBD; ++c) gi += (swk * Rt[c]) * Wg[(DBD + c) * DBD + d];
    float g = 1.f / (1.f + expf(-gi));
    float wr = swk * Rt[d];
    supd[k][d] = (1.f - g) * sl[d] + g * wr;
  }
  __syncthreads();
  for (int k = 0; k < KSLOT; ++k) {
    red[d] = supd[k][d]; __syncthreads();
    for (int st = 64; st > 0; st >>= 1) { if (d < st) red[d] += red[d + st]; __syncthreads(); }
    float mu = red[0] / (float)DBD; __syncthreads();
    float df = supd[k][d] - mu;
    red[d] = df * df; __syncthreads();
    for (int st = 64; st > 0; st >>= 1) { if (d < st) red[d] += red[d + st]; __syncthreads(); }
    float var = red[0] / (float)DBD; __syncthreads();
    Bn[k][d] = df * rsqrtf(var + 1e-5f) * lnw[d] + lnb[d];
  }
  __syncthreads();
  for (int k = 0; k < KSLOT; ++k) {
    float q = 0.f, kk = 0.f, vv = 0.f;
    for (int c = 0; c < DBD; ++c) {
      float bn = Bn[k][c];
      q += bn * WsQ[c * DBD + d]; kk += bn * WsK[c * DBD + d]; vv += bn * WsV[c * DBD + d];
    }
    Qm[k][d] = q; Km[k][d] = kk; Vm[k][d] = vv;
  }
  __syncthreads();
  if (d < KSLOT * KSLOT) {
    int q = d / KSLOT, kx = d % KSLOT;
    float s = 0.f;
    for (int c = 0; c < DBD; ++c) s += Qm[q][c] * Km[kx][c];
    sa[q][kx] = s * scale;
  }
  __syncthreads();
  if (d < KSLOT) {
    float mx = sa[d][0];
    for (int k = 1; k < KSLOT; ++k) mx = fmaxf(mx, sa[d][k]);
    float s = 0.f;
    for (int k = 0; k < KSLOT; ++k) { float e = expf(sa[d][k] - mx); sa[d][k] = e; s += e; }
    for (int k = 0; k < KSLOT; ++k) sa[d][k] /= s;
  }
  __syncthreads();
  {
    float lm = tanhf(lam[d]);
    lm = fminf(0.5f, fmaxf(-0.5f, lm));
    for (int q = 0; q < KSLOT; ++q) {
      float c = 0.f;
      for (int k = 0; k < KSLOT; ++k) c += sa[q][k] * Vm[k][d];
      float sn = supd[q][d] + c * lm;
      snew[q][d] = sn;
      sn_ws[((size_t)b * KSLOT + q) * DBD + d] = sn;
      out_slots[((size_t)b * KSLOT + q) * DBD + d] = sn;
    }
  }
  __syncthreads();
  {
    float p = 0.f;
    for (int k = 0; k < KSLOT; ++k) {
      float df = snew[k][d] - slots[((size_t)b * KSLOT + k) * DBD + d];
      p += df * df;
    }
    red[d] = p; __syncthreads();
    for (int st = 64; st > 0; st >>= 1) { if (d < st) red[d] += red[d + st]; __syncthreads(); }
    if (d == 0) lossP[b] = red[0];
    __syncthreads();
  }
  for (int k = 0; k < KSLOT; ++k) {
    float v = snew[k][d];
    red[d] = v * v; __syncthreads();
    for (int st = 64; st > 0; st >>= 1) { if (d < st) red[d] += red[d + st]; __syncthreads(); }
    if (d == 0) norms[k] = fmaxf(sqrtf(red[0]), 1e-12f);
    __syncthreads();
  }
  if (d < KSLOT * KSLOT) {
    int q = d / KSLOT, kx = d % KSLOT;
    float s = 0.f;
    for (int c = 0; c < DBD; ++c) s += snew[q][c] * snew[kx][c];
    cosv[d] = s / (norms[q] * norms[kx]);
  }
  __syncthreads();
  if (d == 0) {
    float s = 0.f;
    for (int t = 0; t < KSLOT * KSLOT; ++t) {
      if (t / KSLOT != t % KSLOT) { float c = cosv[t]; s += c * c; }
    }
    lossD[b] = s;
  }
}

__global__ void k_loss_final(const float* __restrict__ lossP, const float* __restrict__ lossD,
                             float* __restrict__ out) {
  if (threadIdx.x == 0) {
    float p = 0.f, dv = 0.f;
    for (int b = 0; b < BATCH; ++b) { p += lossP[b]; dv += lossD[b]; }
    out[0] = p / (float)(BATCH * KSLOT * DBD);
    out[1] = dv / (float)(BATCH * (KSLOT * KSLOT - KSLOT));
  }
}

// ---------------- q_mean over T ----------------
__global__ void k_qmean(const float* __restrict__ Xr, float* __restrict__ qm) {
  int b = blockIdx.x;
  for (int d = threadIdx.x; d < DIM; d += 256) {
    float s = 0.f;
    for (int t = 0; t < SEQ; ++t) s += Xr[((size_t)b * SEQ + t) * DIM + d];
    qm[b * DIM + d] = s / (float)SEQ;
  }
}

// ---------------- phi/psi basis, gate, k_r, v_r*gate ----------------
__global__ void k_readprep(const float* __restrict__ qmean, const float* __restrict__ W0,
                           const float* __restrict__ sn, const float* __restrict__ WKr,
                           const float* __restrict__ WVr, const float* __restrict__ rc,
                           float* __restrict__ kr, float* __restrict__ vg) {
  int b = blockIdx.x; int tid = threadIdx.x;
  __shared__ float qm[DIM];
  __shared__ float red[256];
  __shared__ float phmS, psmS, bas1, bas2;
  for (int d = tid; d < DIM; d += 256) qm[d] = qmean[b * DIM + d];
  __syncthreads();
  const float den = PHI_F() - PSI_F();
  float phs = 0.f, pss = 0.f;
  for (int d = tid; d < DIM; d += 256) {
    float w = 0.f;
    for (int c = 0; c < DIM; ++c) w += qm[c] * W0[(size_t)d * DIM + c];
    float ph = (w - PSI_F() * qm[d]) / den;
    float ps = (PHI_F() * qm[d] - w) / den;
    phs += ph * ph; pss += ps * ps;
  }
  red[tid] = phs; __syncthreads();
  for (int st = 128; st > 0; st >>= 1) { if (tid < st) red[tid] += red[tid + st]; __syncthreads(); }
  if (tid == 0) phmS = sqrtf(red[0]);
  __syncthreads();
  red[tid] = pss; __syncthreads();
  for (int st = 128; st > 0; st >>= 1) { if (tid < st) red[tid] += red[tid + st]; __syncthreads(); }
  if (tid == 0) {
    psmS = sqrtf(red[0]);
    float tot = phmS + psmS + 1e-6f;
    bas1 = phmS / tot; bas2 = psmS / tot;
  }
  __syncthreads();
  for (int idx = tid; idx < KSLOT * DBD; idx += 256) {
    int k = idx / DBD, d2 = idx % DBD;
    const float* s = sn + ((size_t)b * KSLOT + k) * DBD;
    float acc = 0.f;
    for (int c = 0; c < DBD; ++c) acc += s[c] * WKr[c * DBD + d2];
    kr[((size_t)b * KSLOT + k) * DBD + d2] = acc;
  }
  for (int idx = tid; idx < KSLOT * DIM; idx += 256) {
    int k = idx / DIM, d = idx % DIM;
    const float* s = sn + ((size_t)b * KSLOT + k) * DBD;
    float vr = 0.f;
    for (int c = 0; c < DBD; ++c) vr += s[c] * WVr[(size_t)c * DIM + d];
    float gt = rc[(size_t)(k * 3 + 0) * DIM + d]
             + rc[(size_t)(k * 3 + 1) * DIM + d] * bas1
             + rc[(size_t)(k * 3 + 2) * DIM + d] * bas2;
    float g = tanhf(gt);
    g = fminf(0.5f, fmaxf(-0.5f, g));
    vg[((size_t)b * KSLOT + k) * DIM + d] = vr * g;
  }
}

// ---------------- read attention: one wave per token ----------------
__global__ void k_readattn(const float* __restrict__ qr, const float* __restrict__ kr,
                           const float* __restrict__ vg, float* __restrict__ z) {
  int b = blockIdx.y;
  int t0 = blockIdx.x * 8;
  int tid = threadIdx.x, wave = tid >> 5, lane = tid & 31;
  __shared__ float krs[KSLOT * DBD];
  __shared__ float vgs[KSLOT * DIM];
  for (int idx = tid; idx < KSLOT * DBD; idx += 256) krs[idx] = kr[(size_t)b * KSLOT * DBD + idx];
  for (int idx = tid; idx < KSLOT * DIM; idx += 256) vgs[idx] = vg[(size_t)b * KSLOT * DIM + idx];
  __syncthreads();
  int t = t0 + wave;
  const float* q = qr + ((size_t)b * SEQ + t) * DBD;
  float q4[4];
  #pragma unroll
  for (int r = 0; r < 4; ++r) q4[r] = q[lane + 32 * r];
  const float scale = rsqrtf((float)DBD);
  float att[KSLOT];
  #pragma unroll
  for (int k = 0; k < KSLOT; ++k) {
    float s = 0.f;
    #pragma unroll
    for (int r = 0; r < 4; ++r) s += q4[r] * krs[k * DBD + lane + 32 * r];
    #pragma unroll
    for (int off = 16; off >= 1; off >>= 1) s += __shfl_xor(s, off, 32);
    att[k] = s * scale;
  }
  float mx = att[0];
  #pragma unroll
  for (int k = 1; k < KSLOT; ++k) mx = fmaxf(mx, att[k]);
  float ssum = 0.f;
  #pragma unroll
  for (int k = 0; k < KSLOT; ++k) { att[k] = expf(att[k] - mx); ssum += att[k]; }
  float inv = 1.f / ssum;
  #pragma unroll
  for (int k = 0; k < KSLOT; ++k) att[k] *= inv;
  float* zo = z + ((size_t)b * SEQ + t) * DIM;
  #pragma unroll
  for (int r = 0; r < 16; ++r) {
    int d = lane + 32 * r;
    float v = 0.f;
    #pragma unroll
    for (int k = 0; k < KSLOT; ++k) v += att[k] * vgs[k * DIM + d];
    zo[d] = v;
  }
}

extern "C" void kernel_launch(void* const* d_in, const int* in_sizes, int n_in,
                              void* d_out, int out_size, void* d_ws, size_t ws_size,
                              hipStream_t stream) {
  (void)in_sizes; (void)n_in; (void)out_size; (void)ws_size;
  const float* X_write  = (const float*)d_in[0];
  const float* X_read   = (const float*)d_in[1];
  const float* slots    = (const float*)d_in[2];
  const float* W0       = (const float*)d_in[3];
  const float* lnww     = (const float*)d_in[4];
  const float* lnwb     = (const float*)d_in[5];
  const float* W_V      = (const float*)d_in[6];
  const float* W_K_slot = (const float*)d_in[7];
  const float* skb      = (const float*)d_in[8];
  const float* Wg       = (const float*)d_in[9];
  const float* Wgb      = (const float*)d_in[10];
  const float* lnsw     = (const float*)d_in[11];
  const float* lnsb     = (const float*)d_in[12];
  const float* WsQ      = (const float*)d_in[13];
  const float* WsK      = (const float*)d_in[14];
  const float* WsV      = (const float*)d_in[15];
  const float* lam      = (const float*)d_in[16];
  const float* WQr      = (const float*)d_in[17];
  const float* WKr      = (const float*)d_in[18];
  const float* WVr      = (const float*)d_in[19];
  const float* rc       = (const float*)d_in[20];
  const float* temps    = (const float*)d_in[21];

  float* out = (float*)d_out;
  float* z = out;
  float* out_slots = out + (size_t)BATCH * SEQ * DIM;
  float* out_loss  = out_slots + (size_t)BATCH * KSLOT * DBD;

  char* ws = (char*)d_ws;
  size_t off = 0;
  auto take = [&](size_t bytes) -> void* {
    void* p = ws + off;
    off += (bytes + 255) & ~(size_t)255;
    return p;
  };
  float*  xn    = (float*)take((size_t)BATCH * SEQ * DIM * 4);
  __bf16* xnt   = (__bf16*)take((size_t)BATCH * SEQ * DIM * 2);
  __bf16* xrt   = (__bf16*)take((size_t)BATCH * SEQ * DIM * 2);
  __bf16* w0t   = (__bf16*)take((size_t)DIM * DIM * 2);
  __bf16* wqt   = (__bf16*)take((size_t)DIM * DBD * 2);
  float*  xw0   = (float*)take((size_t)BATCH * SEQ * DIM * 4);
  float*  sal   = (float*)take((size_t)BATCH * SEQ * WIN * 4);
  float*  tkv   = (float*)take((size_t)BATCH * NTOP * 4);
  int*    tki   = (int*)take((size_t)BATCH * NTOP * 4);
  float*  snws  = (float*)take((size_t)BATCH * KSLOT * DBD * 4);
  float*  qmean = (float*)take((size_t)BATCH * DIM * 4);
  float*  kr    = (float*)take((size_t)BATCH * KSLOT * DBD * 4);
  float*  vg    = (float*)take((size_t)BATCH * KSLOT * DIM * 4);
  float*  qr    = (float*)take((size_t)BATCH * SEQ * DBD * 4);
  float*  lossP = (float*)take((size_t)BATCH * 4);
  float*  lossD = (float*)take((size_t)BATCH * 4);

  k_ln<<<BATCH * SEQ, 256, 0, stream>>>(X_write, lnww, lnwb, xn, xnt);
  k_cvt_tiled<<<2048, 256, 0, stream>>>(X_read, xrt);
  k_tileB<1><<<(32 * KC * 512) / 256, 256, 0, stream>>>(W0, w0t, 32);   // W0^T, N=512
  k_tileB<0><<<(8 * KC * 512) / 256, 256, 0, stream>>>(WQr, wqt, 8);    // W_Q_read, N=128
  k_gemm<<<dim3((BATCH * SEQ) / 64, DIM / 64), 256, 0, stream>>>(xnt, w0t, xw0, DIM);
  k_sal<<<dim3(SEQ / 16, BATCH), 256, 0, stream>>>(xnt, xw0, sal);
  k_topk<<<BATCH, 256, 0, stream>>>(sal, tkv, tki);
  k_slots<<<BATCH, DBD, 0, stream>>>(xn, tkv, tki, slots, W_V, W_K_slot, skb, Wg, Wgb,
                                     lnsw, lnsb, WsQ, WsK, WsV, lam, temps,
                                     snws, out_slots, lossP, lossD);
  k_loss_final<<<1, 32, 0, stream>>>(lossP, lossD, out_loss);
  k_qmean<<<BATCH, 256, 0, stream>>>(X_read, qmean);
  k_readprep<<<BATCH, 256, 0, stream>>>(qmean, W0, snws, WKr, WVr, rc, kr, vg);
  k_gemm<<<dim3((BATCH * SEQ) / 64, DBD / 64), 256, 0, stream>>>(xrt, wqt, qr, DBD);
  k_readattn<<<dim3(SEQ / 8, BATCH), 256, 0, stream>>>(qr, kr, vg, z);
}
